// TriangleAttention_56736517980629
// MI455X (gfx1250) — matrix operand order repaired and checked by reading, not compiled, not run on hardware
//
#include <hip/hip_runtime.h>
#include <hip/hip_bf16.h>

// Triangle attention for MI455X (gfx1250, wave32, WMMA).
// All GEMMs use v_wmma_f32_16x16x32_bf16 (f32 accumulate). Scores + softmax
// live entirely in LDS (128 rows x 257 f32 = 128.5KB < 320KB/WGP); softmax
// repacks P to bf16 in place so the P.V A-fragments are direct dword LDS loads.

#define N_SEQ   256
#define DMODEL  128
#define NHEAD   4
#define HDIM    32
#define NPOS    (N_SEQ * N_SEQ)   // 65536
#define QROWS   128
#define SSTRIDE 257               // padded row stride (bank-conflict free)

typedef __attribute__((ext_vector_type(16))) __bf16       v16bf;
typedef __attribute__((ext_vector_type(8)))  float        v8f;
typedef __attribute__((ext_vector_type(4)))  unsigned int u32x4;
typedef __attribute__((ext_vector_type(2)))  unsigned int u32x2;
typedef __attribute__((ext_vector_type(4)))  float        f32x4;

union Frag16 {
  v16bf v;
  unsigned short us[16];
  unsigned int   ui[8];
  u32x4 q[2];
};

__device__ __forceinline__ unsigned short f2bf(float f) {
  unsigned int u = __float_as_uint(f);
  u += 0x7FFFu + ((u >> 16) & 1u);      // round-to-nearest-even
  return (unsigned short)(u >> 16);
}
__device__ __forceinline__ float bf2f(unsigned short b) {
  return __uint_as_float(((unsigned int)b) << 16);
}
__device__ __forceinline__ float fast_rcp(float x) {
  return __builtin_amdgcn_rcpf(x);      // single v_rcp_f32
}

// A fragment: 16x32 bf16, row m = m0 + lane%16.
// ISA layout: lanes 0-15 elems 0..7 = K0..7, elems 8..15 = K16..23;
//             lanes 16-31 shift K by +8.
__device__ __forceinline__ v16bf load_a_frag(const unsigned short* __restrict__ src,
                                             int row_stride, int m0, int k0, int lane) {
  Frag16 f;
  const unsigned short* p = src + (size_t)(m0 + (lane & 15)) * row_stride
                                + k0 + 8 * (lane >> 4);
  f.q[0] = *(const u32x4*)(p);
  f.q[1] = *(const u32x4*)(p + 16);
  return f.v;
}

// B fragment: 32x16 bf16 read from K-major ("transposed") storage BT[n][k].
// ISA layout: lane n = n0 + lane%16, elems 0..7 = K 8*(lane/16) + 0..7, elems 8..15 = K 16 + 8*(lane/16) + 0..7.
__device__ __forceinline__ v16bf load_b_frag(const unsigned short* __restrict__ bt,
                                             int row_stride, int n0, int k0, int lane) {
  Frag16 f;
  const unsigned short* p = bt + (size_t)(n0 + (lane & 15)) * row_stride
                               + k0 + 8 * (lane >> 4);
  f.q[0] = *(const u32x4*)(p);
  f.q[1] = *(const u32x4*)(p + 16);
  return f.v;
}

// ---------------------------------------------------------------------------
// Kernel 1: transpose weights to bf16 K-major layout.
// WT: [4][128][128] = {Wq,Wk,Wv,Wg} as WT[w][n][k]; WoT: [128][128] = Wo[n][k].
__global__ void prep_weights_kernel(const float* __restrict__ Wq, const float* __restrict__ Wk,
                                    const float* __restrict__ Wv, const float* __restrict__ Wg,
                                    const float* __restrict__ Wo,
                                    unsigned short* __restrict__ WT,
                                    unsigned short* __restrict__ WoT) {
  int tid = blockIdx.x * blockDim.x + threadIdx.x;  // 5 * 16384 threads
  int mat = tid >> 14;
  int i   = tid & 16383;
  int n = i >> 7, k = i & 127;
  const float* src = (mat == 0) ? Wq : (mat == 1) ? Wk : (mat == 2) ? Wv
                   : (mat == 3) ? Wg : Wo;
  unsigned short val = f2bf(src[k * DMODEL + n]);
  if (mat < 4) WT[((size_t)mat * DMODEL + n) * DMODEL + k] = val;
  else         WoT[(size_t)n * DMODEL + k] = val;
}

// ---------------------------------------------------------------------------
// Kernel 2: LayerNorm (one wave per position) + fused bias = z @ Wb.
// zArr: [65536][128] bf16 ; biasT: [H][65536] f32 (bias[h][i*256+j] = zWb[i][j][h]).
__global__ void ln_bias_kernel(const float* __restrict__ x,
                               const float* __restrict__ ln_w,
                               const float* __restrict__ ln_b,
                               const float* __restrict__ Wb,
                               unsigned short* __restrict__ zArr,
                               float* __restrict__ biasT) {
  int lane = threadIdx.x & 31;
  int wave = threadIdx.x >> 5;
  int pos  = blockIdx.x * 8 + wave;
  int c0   = lane * 4;

  f32x4 xv = *(const f32x4*)(x + (size_t)pos * DMODEL + c0);
  float s = xv.x + xv.y + xv.z + xv.w;
  #pragma unroll
  for (int off = 16; off > 0; off >>= 1) s += __shfl_xor(s, off, 32);
  float mu = s * (1.0f / DMODEL);

  f32x4 d;
  d.x = xv.x - mu; d.y = xv.y - mu; d.z = xv.z - mu; d.w = xv.w - mu;
  float vs = d.x * d.x + d.y * d.y + d.z * d.z + d.w * d.w;
  #pragma unroll
  for (int off = 16; off > 0; off >>= 1) vs += __shfl_xor(vs, off, 32);
  float inv = rsqrtf(vs * (1.0f / DMODEL) + 1e-5f);

  float z[4];
  #pragma unroll
  for (int j = 0; j < 4; ++j)
    z[j] = d[j] * inv * ln_w[c0 + j] + ln_b[c0 + j];

  u32x2 pk;
  pk.x = (unsigned int)f2bf(z[0]) | ((unsigned int)f2bf(z[1]) << 16);
  pk.y = (unsigned int)f2bf(z[2]) | ((unsigned int)f2bf(z[3]) << 16);
  *(u32x2*)(zArr + (size_t)pos * DMODEL + c0) = pk;

  float bacc[NHEAD] = {0.f, 0.f, 0.f, 0.f};
  #pragma unroll
  for (int j = 0; j < 4; ++j) {
    int c = c0 + j;
    #pragma unroll
    for (int h = 0; h < NHEAD; ++h) bacc[h] += z[j] * Wb[c * NHEAD + h];
  }
  #pragma unroll
  for (int h = 0; h < NHEAD; ++h) {
    #pragma unroll
    for (int off = 16; off > 0; off >>= 1) bacc[h] += __shfl_xor(bacc[h], off, 32);
  }
  if (lane == 0) {
    #pragma unroll
    for (int h = 0; h < NHEAD; ++h) biasT[(size_t)h * NPOS + pos] = bacc[h];
  }
}

// ---------------------------------------------------------------------------
// Kernel 3: projections z @ {Wq,Wk,Wv,Wg}. One 16x16 tile per wave, K=128.
// q,k,g stored [pos][h*32+c] bf16; v stored transposed vT[(b*4+h)*32+c][jpos].
__global__ void proj_kernel(const unsigned short* __restrict__ zArr,
                            const unsigned short* __restrict__ WT,
                            unsigned short* __restrict__ qArr,
                            unsigned short* __restrict__ kArr,
                            unsigned short* __restrict__ vT,
                            unsigned short* __restrict__ gArr) {
  int lane = threadIdx.x & 31;
  int wave = threadIdx.x >> 5;
  int tile = blockIdx.x * 8 + wave;     // 4096 mtiles x 32 ntiles
  int mt = tile >> 5;
  int nt = tile & 31;
  int w   = nt >> 3;                    // 0=q 1=k 2=v 3=g
  int nc0 = (nt & 7) * 16;              // column base within 128-wide weight
  const unsigned short* bt = WT + (size_t)(w * DMODEL) * DMODEL;

  v8f acc = {};
  #pragma unroll
  for (int ks = 0; ks < 4; ++ks) {
    v16bf a = load_a_frag(zArr, DMODEL, mt * 16, ks * 32, lane);
    v16bf b = load_b_frag(bt, DMODEL, nc0, ks * 32, lane);
    acc = __builtin_amdgcn_wmma_f32_16x16x32_bf16(false, a, false, b,
                                                  (short)0, acc, false, false);
  }

  int nl = lane & 15, hl = lane >> 4;
  int ncol = nc0 + nl;
  #pragma unroll
  for (int vi = 0; vi < 8; ++vi) {
    int m = mt * 16 + vi + 8 * hl;      // global flat position
    float val = acc[vi];
    if (w == 0) {
      qArr[(size_t)m * DMODEL + ncol] = f2bf(val);
    } else if (w == 1) {
      kArr[(size_t)m * DMODEL + ncol] = f2bf(val);
    } else if (w == 2) {
      int b_ = m >> 8, j = m & 255;
      int h = ncol >> 5, c = ncol & 31;
      vT[(size_t)(((b_ * NHEAD + h) * HDIM) + c) * N_SEQ + j] = f2bf(val);
    } else {
      float sg = fast_rcp(1.0f + __expf(-val));   // sigmoid, single v_rcp_f32
      gArr[(size_t)m * DMODEL + ncol] = f2bf(sg);
    }
  }
}

// ---------------------------------------------------------------------------
// Kernel 4: attention. One block per (b, h, half-of-q-rows). 256 threads.
// S (f32) lives in LDS; softmax normalizes + repacks P to bf16 in place so
// the P.V A-fragments are plain dword LDS loads.
__global__ void attn_kernel(const unsigned short* __restrict__ qArr,
                            const unsigned short* __restrict__ kArr,
                            const unsigned short* __restrict__ vT,
                            const unsigned short* __restrict__ gArr,
                            const float* __restrict__ biasT,
                            unsigned short* __restrict__ ogArr) {
  extern __shared__ float S[];          // [QROWS][SSTRIDE]
  int lane = threadIdx.x & 31;
  int wave = threadIdx.x >> 5;
  int bh   = blockIdx.x >> 1;
  int half = blockIdx.x & 1;
  int b = bh >> 2, h = bh & 3;
  const float scale = 0.17677669529663687f;  // 1/sqrt(32)

  const unsigned short* qbase = qArr + (size_t)(b * N_SEQ + half * QROWS) * DMODEL + h * HDIM;
  const unsigned short* kbase = kArr + (size_t)(b * N_SEQ) * DMODEL + h * HDIM;
  const float* brow = biasT + (size_t)h * NPOS;

  // S = Q K^T * scale + bias  (K-dim = HD = 32 -> single WMMA per tile)
  for (int t = wave; t < (QROWS / 16) * 16; t += 8) {
    int qt = t >> 4, kt = t & 15;
    v16bf a  = load_a_frag(qbase, DMODEL, qt * 16, 0, lane);
    v16bf bb = load_b_frag(kbase, DMODEL, kt * 16, 0, lane);
    v8f acc = {};
    acc = __builtin_amdgcn_wmma_f32_16x16x32_bf16(false, a, false, bb,
                                                  (short)0, acc, false, false);
    int nl = lane & 15, hl = lane >> 4;
    #pragma unroll
    for (int vi = 0; vi < 8; ++vi) {
      int m = qt * 16 + vi + 8 * hl;
      int n = kt * 16 + nl;
      int qsec = half * QROWS + m;
      S[m * SSTRIDE + n] = acc[vi] * scale + brow[(size_t)qsec * N_SEQ + n];
    }
  }
  __syncthreads();

  // Row softmax; final pass normalizes and packs P to bf16 in place.
  // Write index t consumes only elements 2t,2t+1 (already read), so the
  // in-place repack is hazard-free for any read-ahead/ write-sink reordering.
  if (threadIdx.x < QROWS) {
    float* row = S + threadIdx.x * SSTRIDE;
    float mx = -1e30f;
    for (int c = 0; c < N_SEQ; ++c) mx = fmaxf(mx, row[c]);
    float sum = 0.f;
    for (int c = 0; c < N_SEQ; ++c) { float e = __expf(row[c] - mx); row[c] = e; sum += e; }
    float inv = fast_rcp(sum);
    unsigned int* row32 = (unsigned int*)row;
    for (int t = 0; t < N_SEQ / 2; ++t) {
      float e0 = row[2 * t] * inv;
      float e1 = row[2 * t + 1] * inv;
      row32[t] = (unsigned int)f2bf(e0) | ((unsigned int)f2bf(e1) << 16);
    }
  }
  __syncthreads();

  // O = P V ; A-fragment = dword loads of packed-bf16 P rows; B from vT.
  const unsigned short* vbase = vT + (size_t)((b * NHEAD + h) * HDIM) * N_SEQ;
  for (int t = wave; t < (QROWS / 16) * 2; t += 8) {
    int mtl = t >> 1, ntl = t & 1;
    int nl = lane & 15, hl = lane >> 4;
    v8f acc = {};
    for (int ks = 0; ks < 8; ++ks) {
      int k0 = ks * 32;
      Frag16 fa;
      const unsigned int* p32 = (const unsigned int*)(S + (size_t)(mtl * 16 + nl) * SSTRIDE)
                              + ((k0 + 8 * hl) >> 1);
      fa.ui[0] = p32[0]; fa.ui[1] = p32[1]; fa.ui[2] = p32[2]; fa.ui[3] = p32[3];
      fa.ui[4] = p32[8]; fa.ui[5] = p32[9]; fa.ui[6] = p32[10]; fa.ui[7] = p32[11];
      v16bf bb = load_b_frag(vbase, N_SEQ, ntl * 16, k0, lane);
      acc = __builtin_amdgcn_wmma_f32_16x16x32_bf16(false, fa.v, false, bb,
                                                    (short)0, acc, false, false);
    }
    #pragma unroll
    for (int vi = 0; vi < 8; ++vi) {
      int m = mtl * 16 + vi + 8 * hl;
      int c = ntl * 16 + nl;
      int qsec = half * QROWS + m;
      size_t pidx = (size_t)(b * N_SEQ + qsec) * DMODEL + h * HDIM + c;
      float gv = bf2f(gArr[pidx]);
      ogArr[pidx] = f2bf(acc[vi] * gv);
    }
  }
}

// ---------------------------------------------------------------------------
// Kernel 5: out = (o*g) @ Wo, f32 result.
__global__ void out_kernel(const unsigned short* __restrict__ ogArr,
                           const unsigned short* __restrict__ WoT,
                           float* __restrict__ out) {
  int lane = threadIdx.x & 31;
  int wave = threadIdx.x >> 5;
  int tile = blockIdx.x * 8 + wave;     // 4096 mtiles x 8 ntiles
  int mt = tile >> 3, nt = tile & 7;
  v8f acc = {};
  #pragma unroll
  for (int ks = 0; ks < 4; ++ks) {
    v16bf a = load_a_frag(ogArr, DMODEL, mt * 16, ks * 32, lane);
    v16bf b = load_b_frag(WoT, DMODEL, nt * 16, ks * 32, lane);
    acc = __builtin_amdgcn_wmma_f32_16x16x32_bf16(false, a, false, b,
                                                  (short)0, acc, false, false);
  }
  int nl = lane & 15, hl = lane >> 4;
  #pragma unroll
  for (int vi = 0; vi < 8; ++vi) {
    int m = mt * 16 + vi + 8 * hl;
    out[(size_t)m * DMODEL + nt * 16 + nl] = acc[vi];
  }
}

// ---------------------------------------------------------------------------
extern "C" void kernel_launch(void* const* d_in, const int* in_sizes, int n_in,
                              void* d_out, int out_size, void* d_ws, size_t ws_size,
                              hipStream_t stream) {
  (void)in_sizes; (void)n_in; (void)out_size; (void)ws_size;
  const float* x    = (const float*)d_in[0];
  const float* ln_w = (const float*)d_in[1];
  const float* ln_b = (const float*)d_in[2];
  const float* Wb   = (const float*)d_in[3];
  const float* Wq   = (const float*)d_in[4];
  const float* Wk   = (const float*)d_in[5];
  const float* Wv   = (const float*)d_in[6];
  const float* Wg   = (const float*)d_in[7];
  const float* Wo   = (const float*)d_in[8];
  float* out = (float*)d_out;

  char* ws = (char*)d_ws;
  size_t off = 0;
  auto alloc = [&](size_t bytes) -> char* {
    char* p = ws + off;
    off += (bytes + 255) & ~(size_t)255;
    return p;
  };
  unsigned short* zArr  = (unsigned short*)alloc((size_t)NPOS * DMODEL * 2);
  unsigned short* qArr  = (unsigned short*)alloc((size_t)NPOS * DMODEL * 2);
  unsigned short* kArr  = (unsigned short*)alloc((size_t)NPOS * DMODEL * 2);
  unsigned short* vTA   = (unsigned short*)alloc((size_t)NPOS * DMODEL * 2);
  unsigned short* gArr  = (unsigned short*)alloc((size_t)NPOS * DMODEL * 2);
  unsigned short* ogArr = (unsigned short*)alloc((size_t)NPOS * DMODEL * 2);
  float*          biasT = (float*)alloc((size_t)NHEAD * NPOS * 4);
  unsigned short* WT    = (unsigned short*)alloc((size_t)4 * DMODEL * DMODEL * 2);
  unsigned short* WoT   = (unsigned short*)alloc((size_t)DMODEL * DMODEL * 2);

  const int smem = QROWS * SSTRIDE * (int)sizeof(float);  // 131,584 B (<320KB/WGP)
  (void)hipFuncSetAttribute((const void*)attn_kernel,
                            hipFuncAttributeMaxDynamicSharedMemorySize, smem);

  prep_weights_kernel<<<(5 * DMODEL * DMODEL) / 256, 256, 0, stream>>>(
      Wq, Wk, Wv, Wg, Wo, WT, WoT);
  ln_bias_kernel<<<NPOS / 8, 256, 0, stream>>>(x, ln_w, ln_b, Wb, zArr, biasT);
  proj_kernel<<<(NPOS / 16) * 32 / 8, 256, 0, stream>>>(zArr, WT, qArr, kArr, vTA, gArr);
  attn_kernel<<<N_SEQ * NHEAD * 2, 256, smem, stream>>>(qArr, kArr, vTA, gArr, biasT, ogArr);
  out_kernel<<<(NPOS / 16) * 8 / 8, 256, 0, stream>>>(ogArr, WoT, out);
}
